// MixtureOfExperts_515396075673
// MI455X (gfx1250) — matrix operand order repaired and checked by reading, not compiled, not run on hardware
//
#include <hip/hip_runtime.h>
#include <hip/hip_bf16.h>
#include <cstdint>

#define DIM  1024
#define HID  2048
#define NE   8
#define NT   4096
#define TPB  256
#define MT   32            // tokens per expert tile
#define TPB2 512           // threads per block, expert kernel (16 waves)
#define NW   (TPB2 / 32)

typedef __attribute__((ext_vector_type(16))) __bf16        v16bf;
typedef __attribute__((ext_vector_type(8)))  float         v8f;
typedef __attribute__((ext_vector_type(8)))  unsigned int  v8u;
typedef __attribute__((ext_vector_type(4)))  int           v4i;

#if defined(__has_builtin)
#if __has_builtin(__builtin_amdgcn_global_load_async_to_lds_b128) && \
    __has_builtin(__builtin_amdgcn_s_wait_asynccnt)
#define HAS_ASYNC 1
#endif
#endif
#ifndef HAS_ASYNC
#define HAS_ASYNC 0
#endif

typedef __attribute__((address_space(1))) v4i as1_v4i;   // global int4
typedef __attribute__((address_space(3))) v4i as3_v4i;   // LDS int4

static __device__ __forceinline__ unsigned short f2bf(float f) {
    unsigned int u = __float_as_uint(f);
    u += 0x7fffu + ((u >> 16) & 1u);          // round-to-nearest-even
    return (unsigned short)(u >> 16);
}

// Build a 16xbf16 WMMA fragment from two 16-byte chunks.
static __device__ __forceinline__ v16bf load_frag(const unsigned short* p0,
                                                  const unsigned short* p1) {
    uint4 a = *(const uint4*)p0;
    uint4 b = *(const uint4*)p1;
    v8u u;
    u[0] = a.x; u[1] = a.y; u[2] = a.z; u[3] = a.w;
    u[4] = b.x; u[5] = b.y; u[6] = b.z; u[7] = b.w;
    return __builtin_bit_cast(v16bf, u);
}

// ---------------------------------------------------------------------------
__global__ void init_cnt_kernel(int* cnt) {
    if (threadIdx.x < NE) cnt[threadIdx.x] = 0;
}

// ---------------------------------------------------------------------------
// Router: one wave32 per token -> top-2 experts, renormalized gates, lists.
__global__ void __launch_bounds__(TPB) router_kernel(
        const float* __restrict__ x, const float* __restrict__ Wr,
        float* __restrict__ topw, int* __restrict__ list, int* __restrict__ cnt) {
    int wid  = (blockIdx.x * TPB + threadIdx.x) >> 5;   // token id
    int lane = threadIdx.x & 31;
    if (wid >= NT) return;

    const float* xr = x + (size_t)wid * DIM;
    float acc[NE];
#pragma unroll
    for (int e = 0; e < NE; ++e) acc[e] = 0.f;
    for (int d = lane; d < DIM; d += 32) {
        float xv = xr[d];
        const float* wr = Wr + (size_t)d * NE;
#pragma unroll
        for (int e = 0; e < NE; ++e) acc[e] = fmaf(xv, wr[e], acc[e]);
    }
#pragma unroll
    for (int off = 16; off > 0; off >>= 1) {
#pragma unroll
        for (int e = 0; e < NE; ++e) acc[e] += __shfl_down(acc[e], off);
    }
    if (lane == 0) {
        int e0 = 0; float l0 = acc[0];
#pragma unroll
        for (int e = 1; e < NE; ++e) if (acc[e] > l0) { l0 = acc[e]; e0 = e; }
        int e1 = -1; float l1 = -3.4e38f;
#pragma unroll
        for (int e = 0; e < NE; ++e) if (e != e0 && acc[e] > l1) { l1 = acc[e]; e1 = e; }
        float w0 = 1.f / (1.f + __expf(l1 - l0));
        float w1 = 1.f - w0;
        topw[wid * 2 + 0] = w0;
        topw[wid * 2 + 1] = w1;
        int p0 = atomicAdd(&cnt[e0], 1); list[e0 * NT + p0] = wid * 2 + 0;
        int p1 = atomicAdd(&cnt[e1], 1); list[e1 * NT + p1] = wid * 2 + 1;
    }
}

// ---------------------------------------------------------------------------
// f32 [E][In][Out] -> bf16 K-major [E][Out][In] (tiled transpose).
__global__ void convert_transpose_kernel(const float* __restrict__ src,
                                         unsigned short* __restrict__ dst,
                                         int In, int Out) {
    __shared__ unsigned short tile[32][33];
    int e = blockIdx.z;
    const float*    s = src + (size_t)e * In * Out;
    unsigned short* d = dst + (size_t)e * In * Out;
    int in0 = blockIdx.x * 32, out0 = blockIdx.y * 32;
    int tx = threadIdx.x, ty = threadIdx.y;     // block (32, 8)
#pragma unroll
    for (int j = 0; j < 32; j += 8)
        tile[ty + j][tx] = f2bf(s[(size_t)(in0 + ty + j) * Out + out0 + tx]);
    __syncthreads();
#pragma unroll
    for (int j = 0; j < 32; j += 8)
        d[(size_t)(out0 + ty + j) * In + in0 + tx] = tile[tx][ty + j];
}

// ---------------------------------------------------------------------------
// x f32 -> bf16 (row layout unchanged); enables async LDS gather in stage 3.
__global__ void convert_x_kernel(const float* __restrict__ x,
                                 unsigned short* __restrict__ xb) {
    int i = blockIdx.x * blockDim.x + threadIdx.x;   // one float4 per thread
    float4 v = ((const float4*)x)[i];
    uint2 o;
    o.x = (unsigned)f2bf(v.x) | ((unsigned)f2bf(v.y) << 16);
    o.y = (unsigned)f2bf(v.z) | ((unsigned)f2bf(v.w) << 16);
    ((uint2*)xb)[i] = o;
}

// ---------------------------------------------------------------------------
// Fused SwiGLU expert: block = 16 waves = one 32-token tile of one expert.
__global__ void __launch_bounds__(TPB2) moe_expert_kernel(
        const unsigned short* __restrict__ xb,    // [NT][DIM] bf16
        const unsigned short* __restrict__ W1t,   // [E][HID][DIM] bf16 K-major
        const unsigned short* __restrict__ W3t,   // [E][HID][DIM]
        const unsigned short* __restrict__ W2t,   // [E][DIM][HID]
        const float* __restrict__ topw,
        const int*  __restrict__ list,
        const int*  __restrict__ cnt,
        float* __restrict__ Yp) {                 // [NT*2][DIM]
    __shared__ __align__(16) unsigned short Xs[MT][DIM];   //  64 KB
    __shared__ __align__(16) unsigned short Hs[MT][HID];   // 128 KB
    __shared__ int   s_slot[MT];
    __shared__ float s_w[MT];
    __shared__ int   s_tok[MT];

    const int e  = blockIdx.y;
    const int nv = cnt[e];
    const int m0 = blockIdx.x * MT;
    if (m0 >= nv) return;

    const int tid  = threadIdx.x;
    const int lane = tid & 31;
    const int wv   = tid >> 5;        // wave 0..15
    const int lm   = lane & 15;
    const int hi   = lane >> 4;       // half-wave select

    if (tid < MT) {
        int idx = m0 + tid;
        if (idx < nv) {
            int enc = list[e * NT + idx];
            s_slot[tid] = enc;
            s_w[tid]    = topw[enc];
            s_tok[tid]  = enc >> 1;
        } else {
            s_slot[tid] = -1; s_w[tid] = 0.f; s_tok[tid] = 0;
        }
    }
    __syncthreads();

    // Gather the 32 token rows (bf16) into LDS.
#if HAS_ASYNC
    for (int c = tid; c < MT * DIM / 8; c += TPB2) {       // 16B chunks
        int r   = c >> 7;                                  // DIM/8 = 128 chunks/row
        int off = (c & 127) * 8;
        __builtin_amdgcn_global_load_async_to_lds_b128(
            (as1_v4i*)(void*)(const_cast<unsigned short*>(xb) +
                              (size_t)s_tok[r] * DIM + off),
            (as3_v4i*)(void*)&Xs[r][off], 0, 0);
    }
    __builtin_amdgcn_s_wait_asynccnt(0);
    __syncthreads();
#else
    for (int c = tid; c < MT * DIM / 8; c += TPB2) {
        int r   = c >> 7;
        int off = (c & 127) * 8;
        *(uint4*)&Xs[r][off] =
            *(const uint4*)(xb + (size_t)s_tok[r] * DIM + off);
    }
    __syncthreads();
#endif

    const unsigned short* w1e = W1t + (size_t)e * HID * DIM;
    const unsigned short* w3e = W3t + (size_t)e * HID * DIM;
    const unsigned short* a0p = &Xs[lm][hi * 8];
    const unsigned short* a1p = &Xs[16 + lm][hi * 8];

    // Loop 1: wave wv owns H columns [wv*128, wv*128+128): 8 n-tiles.
    for (int nt = 0; nt < HID / (16 * NW); ++nt) {
        int n0 = wv * (HID / NW) + nt * 16;
        v8f c10 = {}, c11 = {}, c30 = {}, c31 = {};
        const unsigned short* b1 = w1e + (size_t)(n0 + lm) * DIM + hi * 8;
        const unsigned short* b3 = w3e + (size_t)(n0 + lm) * DIM + hi * 8;
        for (int k = 0; k < DIM; k += 32) {
            v16bf af0 = load_frag(a0p + k, a0p + k + 16);  // rows 0-15
            v16bf af1 = load_frag(a1p + k, a1p + k + 16);  // rows 16-31
            v16bf b1f = load_frag(b1 + k, b1 + k + 16);
            v16bf b3f = load_frag(b3 + k, b3 + k + 16);
            c10 = __builtin_amdgcn_wmma_f32_16x16x32_bf16(false, af0, false, b1f,
                                                          (short)0, c10, false, false);
            c11 = __builtin_amdgcn_wmma_f32_16x16x32_bf16(false, af1, false, b1f,
                                                          (short)0, c11, false, false);
            c30 = __builtin_amdgcn_wmma_f32_16x16x32_bf16(false, af0, false, b3f,
                                                          (short)0, c30, false, false);
            c31 = __builtin_amdgcn_wmma_f32_16x16x32_bf16(false, af1, false, b3f,
                                                          (short)0, c31, false, false);
        }
#pragma unroll
        for (int i = 0; i < 8; ++i) {
            float g0 = c10[i], g1 = c11[i];
            float h0 = g0 * __builtin_amdgcn_rcpf(1.f + __expf(-g0)) * c30[i];
            float h1 = g1 * __builtin_amdgcn_rcpf(1.f + __expf(-g1)) * c31[i];
            Hs[i + hi * 8][n0 + lm]      = f2bf(h0);
            Hs[16 + i + hi * 8][n0 + lm] = f2bf(h1);
        }
    }
    __syncthreads();

    // Loop 2: Y = H @ W2. Wave wv owns output cols [wv*64, wv*64+64): 4 n-tiles.
    const unsigned short* w2e = W2t + (size_t)e * DIM * HID;
    const unsigned short* h0p = &Hs[lm][hi * 8];
    const unsigned short* h1p = &Hs[16 + lm][hi * 8];
    for (int nt = 0; nt < DIM / (16 * NW); ++nt) {
        int n0 = wv * (DIM / NW) + nt * 16;
        v8f c0 = {}, c1 = {};
        const unsigned short* b2 = w2e + (size_t)(n0 + lm) * HID + hi * 8;
        for (int k = 0; k < HID; k += 32) {
            v16bf af0 = load_frag(h0p + k, h0p + k + 16);
            v16bf af1 = load_frag(h1p + k, h1p + k + 16);
            v16bf bf  = load_frag(b2 + k, b2 + k + 16);
            c0 = __builtin_amdgcn_wmma_f32_16x16x32_bf16(false, af0, false, bf,
                                                         (short)0, c0, false, false);
            c1 = __builtin_amdgcn_wmma_f32_16x16x32_bf16(false, af1, false, bf,
                                                         (short)0, c1, false, false);
        }
#pragma unroll
        for (int i = 0; i < 8; ++i) {
            int m = i + hi * 8;
            int s0 = s_slot[m], s1 = s_slot[16 + m];
            if (s0 >= 0) Yp[(size_t)s0 * DIM + n0 + lm] = s_w[m]      * c0[i];
            if (s1 >= 0) Yp[(size_t)s1 * DIM + n0 + lm] = s_w[16 + m] * c1[i];
        }
    }
}

// ---------------------------------------------------------------------------
// out[t] = Yp[2t] + Yp[2t+1] (fixed order -> deterministic), float4-wide.
__global__ void combine_kernel(const float* __restrict__ Yp, float* __restrict__ out) {
    int i = blockIdx.x * blockDim.x + threadIdx.x;      // one float4
    if (i >= NT * DIM / 4) return;
    int t  = i >> 8;                                    // DIM/4 = 256 float4 per row
    int d4 = i & 255;
    const float4* a = (const float4*)(Yp + (size_t)(2 * t) * DIM) + d4;
    const float4* b = (const float4*)(Yp + (size_t)(2 * t + 1) * DIM) + d4;
    float4 va = *a, vb = *b;
    float4 r; r.x = va.x + vb.x; r.y = va.y + vb.y; r.z = va.z + vb.z; r.w = va.w + vb.w;
    ((float4*)out)[i] = r;
}

// ---------------------------------------------------------------------------
extern "C" void kernel_launch(void* const* d_in, const int* in_sizes, int n_in,
                              void* d_out, int out_size, void* d_ws, size_t ws_size,
                              hipStream_t stream) {
    (void)in_sizes; (void)n_in; (void)out_size; (void)ws_size;
    const float* x  = (const float*)d_in[0];
    const float* Wr = (const float*)d_in[1];
    const float* W1 = (const float*)d_in[2];
    const float* W2 = (const float*)d_in[3];
    const float* W3 = (const float*)d_in[4];
    float* out = (float*)d_out;

    char* base = (char*)d_ws;
    const size_t SZ_W = (size_t)NE * HID * DIM * sizeof(unsigned short); // 33.5 MB each
    unsigned short* W1t = (unsigned short*)(base);
    unsigned short* W3t = (unsigned short*)(base + SZ_W);
    unsigned short* W2t = (unsigned short*)(base + 2 * SZ_W);
    float* Yp   = (float*)(base + 3 * SZ_W);                       // [NT*2][DIM] f32
    unsigned short* xb = (unsigned short*)((char*)Yp + (size_t)NT * 2 * DIM * sizeof(float));
    float* topw = (float*)((char*)xb + (size_t)NT * DIM * sizeof(unsigned short));
    int*   list = (int*)((char*)topw + (size_t)NT * 2 * sizeof(float));
    int*   cnt  = (int*)((char*)list + (size_t)NE * NT * sizeof(int));

    init_cnt_kernel<<<1, 32, 0, stream>>>(cnt);

    dim3 tb(32, 8);
    convert_transpose_kernel<<<dim3(DIM / 32, HID / 32, NE), tb, 0, stream>>>(W1, W1t, DIM, HID);
    convert_transpose_kernel<<<dim3(DIM / 32, HID / 32, NE), tb, 0, stream>>>(W3, W3t, DIM, HID);
    convert_transpose_kernel<<<dim3(HID / 32, DIM / 32, NE), tb, 0, stream>>>(W2, W2t, HID, DIM);

    convert_x_kernel<<<(NT * DIM / 4) / TPB, TPB, 0, stream>>>(x, xb);

    router_kernel<<<NT / 8, TPB, 0, stream>>>(x, Wr, topw, list, cnt);

    moe_expert_kernel<<<dim3(NT / MT, NE), TPB2, 0, stream>>>(
        xb, W1t, W3t, W2t, topw, list, cnt, Yp);

    combine_kernel<<<(NT * DIM / 4 + TPB - 1) / TPB, TPB, 0, stream>>>(Yp, out);
}
